// FancyBlock_58755152609782
// MI455X (gfx1250) — matrix operand order repaired and checked by reading, not compiled, not run on hardware
//
#include <hip/hip_runtime.h>
#include <cstdint>
#include <cstddef>

typedef __attribute__((ext_vector_type(16))) __bf16 v16bf;
typedef __attribute__((ext_vector_type(8)))  float  v8f;
typedef __attribute__((ext_vector_type(4)))  unsigned int u32x4;

#define C_DIM   768
#define NH      12
#define HD      64
#define WSZ     14
#define NTOK    196     // tokens per window
#define NP      224     // padded token count (7 * 32)
#define BW      256     // number of windows
#define WHTOT   3072    // BW * NH
#define MROWS   50176   // BW * NTOK  (divisible by 32)
#define QKV_N   2304
#define KSTEPS  (C_DIM / 32)   // 24

// ---------- bf16 helpers ----------
static __device__ inline unsigned short f2bf(float x) {
  union { float f; unsigned int u; } c; c.f = x;
  unsigned int r = c.u + 0x7FFFu + ((c.u >> 16) & 1u);
  return (unsigned short)(r >> 16);
}
static __device__ inline float bf2f(unsigned short x) {
  union { unsigned int u; float f; } c; c.u = ((unsigned int)x) << 16;
  return c.f;
}

// ---------- WMMA wrapper (CDNA5 wave32) ----------
static __device__ inline v8f wmma_bf16(v16bf a, v16bf b, v8f c) {
  return __builtin_amdgcn_wmma_f32_16x16x32_bf16(
      /*neg_a=*/false, a, /*neg_b=*/false, b,
      /*c_mod=*/(short)0, c, /*reuse_a=*/false, /*reuse_b=*/false);
}

// A fragment: 16(M) x 32(K) bf16.  p -> element (row0, k0), row stride = ld elems.
// Lane l (l<16): row l, K = {0..7, 16..23}; lane l+16: row l, K = {8..15, 24..31}.
static __device__ inline v16bf load_a(const unsigned short* p, int ld) {
  int lane = threadIdx.x & 31;
  int r = lane & 15, koff = (lane >> 4) * 8;
  const unsigned short* rp = p + r * ld + koff;
  union { v16bf v; u32x4 u[2]; } f;
  f.u[0] = *(const u32x4*)(rp);
  f.u[1] = *(const u32x4*)(rp + 16);
  return f.v;
}

// B fragment: 32(K) x 16(N) bf16, memory holds B^T row-major (row n = K contiguous).
// Lane l (l<16): column l, K = {0..7, 16..23}; lane l+16: column l, K = {8..15, 24..31}.
static __device__ inline v16bf load_bT(const unsigned short* p, int ld) {
  int lane = threadIdx.x & 31;
  int n = lane & 15, kb = (lane >> 4) * 8;
  const unsigned short* rp = p + n * ld + kb;
  union { v16bf v; u32x4 u[2]; } f;
  f.u[0] = *(const u32x4*)(rp);
  f.u[1] = *(const u32x4*)(rp + 16);
  return f.v;
}

// ---------- K1: convert weights to bf16 ----------
__global__ __launch_bounds__(256) void conv_w_kernel(const float* wq, const float* wp,
                                                     unsigned short* oq, unsigned short* op) {
  size_t i = (size_t)blockIdx.x * 256 + threadIdx.x;
  const size_t nq = (size_t)QKV_N * C_DIM;
  const size_t np = (size_t)C_DIM * C_DIM;
  if (i < nq) oq[i] = f2bf(wq[i]);
  else if (i < nq + np) op[i - nq] = f2bf(wp[i - nq]);
}

// ---------- K2: window partition x -> bf16 (Bw, N, C) ----------
__global__ __launch_bounds__(256) void winpart_kernel(const float* x, unsigned short* Xw) {
  size_t i = (size_t)blockIdx.x * 256 + threadIdx.x;
  if (i >= (size_t)MROWS * C_DIM) return;
  int c = (int)(i % C_DIM);
  int m = (int)(i / C_DIM);
  int wb = m / NTOK, tok = m % NTOK;
  int b = wb >> 4, wy = (wb >> 2) & 3, wx = wb & 3;
  int iy = tok / WSZ, ix = tok % WSZ;
  size_t src = (((size_t)b * 56 + (wy * WSZ + iy)) * 56 + (wx * WSZ + ix)) * C_DIM + c;
  Xw[i] = f2bf(x[src]);
}

// ---------- K3: zero padded rows of Q/K (tok 196..223) and pad cols of V^T ----------
__global__ __launch_bounds__(256) void zeropad_kernel(unsigned short* Qb, unsigned short* Kb,
                                                      unsigned short* Vb) {
  int i = blockIdx.x * 256 + threadIdx.x;
  const int per = (NP - NTOK) * HD;            // 28*64
  if (i >= WHTOT * per) return;
  int wh = i / per, rem = i % per;
  int row = NTOK + rem / HD, d = rem % HD;
  Qb[((size_t)wh * NP + row) * HD + d] = 0;
  Kb[((size_t)wh * NP + row) * HD + d] = 0;
  int dd = rem / (NP - NTOK), tok = NTOK + rem % (NP - NTOK);
  Vb[((size_t)wh * HD + dd) * NP + tok] = 0;
}

// ---------- K4: QKV GEMM (M=50176, N=2304, K=768), 32x64 tile/wave, double-buffered ----------
__global__ __launch_bounds__(256) void qkv_gemm_kernel(const unsigned short* __restrict__ Xw,
                                                       const unsigned short* __restrict__ Wq,
                                                       const float* __restrict__ bias,
                                                       unsigned short* __restrict__ Qb,
                                                       unsigned short* __restrict__ Kb,
                                                       unsigned short* __restrict__ Vb) {
  int wid  = (blockIdx.x * 256 + threadIdx.x) >> 5;
  int lane = threadIdx.x & 31;
  const int NT64 = QKV_N / 64;                        // 36
  const int MT   = MROWS / 32;                        // 1568
  if (wid >= MT * NT64) return;
  int mt = wid / NT64, nt = wid % NT64;

  const unsigned short* A0 = Xw + (size_t)mt * 32 * C_DIM;
  const unsigned short* A1 = A0 + (size_t)16 * C_DIM;
  const unsigned short* Bp = Wq + (size_t)nt * 64 * C_DIM;

  v8f acc[2][4];
#pragma unroll
  for (int mi = 0; mi < 2; ++mi)
#pragma unroll
    for (int t = 0; t < 4; ++t) acc[mi][t] = v8f{};

  // prologue loads (k = 0)
  v16bf a0 = load_a(A0, C_DIM), a1 = load_a(A1, C_DIM);
  v16bf b0 = load_bT(Bp + 0 * 16 * C_DIM, C_DIM);
  v16bf b1 = load_bT(Bp + 1 * 16 * C_DIM, C_DIM);
  v16bf b2 = load_bT(Bp + 2 * 16 * C_DIM, C_DIM);
  v16bf b3 = load_bT(Bp + 3 * 16 * C_DIM, C_DIM);

  for (int ks = 1; ks < KSTEPS; ++ks) {
    int k = ks * 32;
    // prefetch next k-step while current WMMAs run
    v16bf a0n = load_a(A0 + k, C_DIM), a1n = load_a(A1 + k, C_DIM);
    v16bf b0n = load_bT(Bp + 0 * 16 * C_DIM + k, C_DIM);
    v16bf b1n = load_bT(Bp + 1 * 16 * C_DIM + k, C_DIM);
    v16bf b2n = load_bT(Bp + 2 * 16 * C_DIM + k, C_DIM);
    v16bf b3n = load_bT(Bp + 3 * 16 * C_DIM + k, C_DIM);
    acc[0][0] = wmma_bf16(a0, b0, acc[0][0]);
    acc[1][0] = wmma_bf16(a1, b0, acc[1][0]);
    acc[0][1] = wmma_bf16(a0, b1, acc[0][1]);
    acc[1][1] = wmma_bf16(a1, b1, acc[1][1]);
    acc[0][2] = wmma_bf16(a0, b2, acc[0][2]);
    acc[1][2] = wmma_bf16(a1, b2, acc[1][2]);
    acc[0][3] = wmma_bf16(a0, b3, acc[0][3]);
    acc[1][3] = wmma_bf16(a1, b3, acc[1][3]);
    a0 = a0n; a1 = a1n; b0 = b0n; b1 = b1n; b2 = b2n; b3 = b3n;
  }
  acc[0][0] = wmma_bf16(a0, b0, acc[0][0]);
  acc[1][0] = wmma_bf16(a1, b0, acc[1][0]);
  acc[0][1] = wmma_bf16(a0, b1, acc[0][1]);
  acc[1][1] = wmma_bf16(a1, b1, acc[1][1]);
  acc[0][2] = wmma_bf16(a0, b2, acc[0][2]);
  acc[1][2] = wmma_bf16(a1, b2, acc[1][2]);
  acc[0][3] = wmma_bf16(a0, b3, acc[0][3]);
  acc[1][3] = wmma_bf16(a1, b3, acc[1][3]);

  int nloc = lane & 15, mh = (lane >> 4) * 8;
#pragma unroll
  for (int mi = 0; mi < 2; ++mi) {
#pragma unroll
    for (int t = 0; t < 4; ++t) {
      int n = nt * 64 + t * 16 + nloc;
      int which = n / C_DIM, c = n % C_DIM;
      int head = c >> 6, d = c & 63;
      float bv = bias[n];
#pragma unroll
      for (int r = 0; r < 8; ++r) {
        int m = mt * 32 + mi * 16 + r + mh;
        int wb = m / NTOK, tok = m % NTOK;
        int wh = wb * NH + head;
        float val = acc[mi][t][r] + bv;
        if (which == 0)       Qb[((size_t)wh * NP + tok) * HD + d] = f2bf(val * 0.125f);
        else if (which == 1)  Kb[((size_t)wh * NP + tok) * HD + d] = f2bf(val);
        else                  Vb[((size_t)wh * HD + d) * NP + tok] = f2bf(val);
      }
    }
  }
}

// ---------- K5: attention per (window-head, q-tile) ----------
__global__ __launch_bounds__(32) void attn_kernel(const unsigned short* __restrict__ Q,
                                                  const unsigned short* __restrict__ K,
                                                  const unsigned short* __restrict__ Vt,
                                                  const float* __restrict__ rph,
                                                  const float* __restrict__ rpw,
                                                  unsigned short* __restrict__ Out) {
  int wh = blockIdx.x;          // 0..3071
  int qt = blockIdx.y;          // 0..12
  int lane = threadIdx.x;

  __shared__ float S[16][NP - 16];      // scores, cols 0..207
  __shared__ float relh[16][16];
  __shared__ float relw[16][16];
  __shared__ unsigned short P[16][NP];  // softmaxed probs (bf16), cols 208..223 zero

  const unsigned short* Qw = Q  + (size_t)wh * NP * HD;
  const unsigned short* Kw = K  + (size_t)wh * NP * HD;
  const unsigned short* Vw = Vt + (size_t)wh * HD * NP;

  // Q A-fragments (k = 0..31 and 32..63)
  v16bf aq0 = load_a(Qw + (size_t)qt * 16 * HD + 0,  HD);
  v16bf aq1 = load_a(Qw + (size_t)qt * 16 * HD + 32, HD);

  // decomposed relative position bias: relh[r][h2], relw[r][w2] (x8 undoes q pre-scale)
  for (int e = lane; e < 16 * 28; e += 32) {
    int r = e / 28, j = e % 28;
    int tok = qt * 16 + r;
    float acc = 0.0f;
    if (tok < NTOK) {
      int h1 = tok / WSZ, w1 = tok % WSZ;
      const float* trow;
      if (j < WSZ) trow = rph + (size_t)(h1 - j + WSZ - 1) * HD;
      else         trow = rpw + (size_t)(w1 - (j - WSZ) + WSZ - 1) * HD;
      const unsigned short* qrow = Qw + (size_t)tok * HD;
      for (int d = 0; d < HD; ++d) acc += bf2f(qrow[d]) * trow[d];
      acc *= 8.0f;
    }
    if (j < WSZ) relh[r][j] = acc; else relw[r][j - WSZ] = acc;
  }
  __syncthreads();

  // S = Q K^T + bias, double-buffered over 13 key tiles (cols >=196 masked)
  {
    v16bf kb0 = load_bT(Kw + 0,  HD);
    v16bf kb1 = load_bT(Kw + 32, HD);
    for (int kt = 0; kt < 13; ++kt) {
      v16bf kb0n = kb0, kb1n = kb1;
      if (kt < 12) {
        kb0n = load_bT(Kw + (size_t)(kt + 1) * 16 * HD + 0,  HD);
        kb1n = load_bT(Kw + (size_t)(kt + 1) * 16 * HD + 32, HD);
      }
      v8f acc = v8f{};
      acc = wmma_bf16(aq0, kb0, acc);
      acc = wmma_bf16(aq1, kb1, acc);
      int n = lane & 15, mh = (lane >> 4) * 8;
      int col = kt * 16 + n;
      int h2 = col / WSZ, w2 = col % WSZ;
      bool valid = col < NTOK;
#pragma unroll
      for (int r = 0; r < 8; ++r) {
        int m = r + mh;
        S[m][col] = valid ? (acc[r] + relh[m][h2] + relw[m][w2]) : -1e30f;
      }
      kb0 = kb0n; kb1 = kb1n;
    }
  }
  __syncthreads();

  // row softmax (lanes 0..15 each own one row)
  if (lane < 16) {
    float mx = -1e30f;
    for (int c = 0; c < NP - 16; ++c) mx = fmaxf(mx, S[lane][c]);
    float sum = 0.0f;
    for (int c = 0; c < NP - 16; ++c) {
      float e = __expf(S[lane][c] - mx);
      S[lane][c] = e; sum += e;
    }
    float inv = 1.0f / sum;
    for (int c = 0; c < NP - 16; ++c) P[lane][c] = f2bf(S[lane][c] * inv);
    for (int c = NP - 16; c < NP; ++c) P[lane][c] = 0;
  }
  __syncthreads();

  // O = P @ V  (K = 224 -> 7 k-steps; V stored transposed), double-buffered B frags
  v8f o0 = v8f{}, o1 = v8f{}, o2 = v8f{}, o3 = v8f{};
  int rr = lane & 15, koff = (lane >> 4) * 8;
  v16bf vb0 = load_bT(Vw + (size_t)0 * 16 * NP, NP);
  v16bf vb1 = load_bT(Vw + (size_t)1 * 16 * NP, NP);
  v16bf vb2 = load_bT(Vw + (size_t)2 * 16 * NP, NP);
  v16bf vb3 = load_bT(Vw + (size_t)3 * 16 * NP, NP);
  for (int kt = 0; kt < 7; ++kt) {
    int kb = kt * 32;
    v16bf vb0n = vb0, vb1n = vb1, vb2n = vb2, vb3n = vb3;
    if (kt < 6) {
      vb0n = load_bT(Vw + (size_t)0 * 16 * NP + kb + 32, NP);
      vb1n = load_bT(Vw + (size_t)1 * 16 * NP + kb + 32, NP);
      vb2n = load_bT(Vw + (size_t)2 * 16 * NP + kb + 32, NP);
      vb3n = load_bT(Vw + (size_t)3 * 16 * NP + kb + 32, NP);
    }
    union { v16bf v; unsigned short s[16]; } f;
#pragma unroll
    for (int i = 0; i < 8; ++i) f.s[i]     = P[rr][kb + koff + i];
#pragma unroll
    for (int i = 0; i < 8; ++i) f.s[8 + i] = P[rr][kb + koff + 16 + i];
    o0 = wmma_bf16(f.v, vb0, o0);
    o1 = wmma_bf16(f.v, vb1, o1);
    o2 = wmma_bf16(f.v, vb2, o2);
    o3 = wmma_bf16(f.v, vb3, o3);
    vb0 = vb0n; vb1 = vb1n; vb2 = vb2n; vb3 = vb3n;
  }

  // store to AttnOut bf16 (Bw, N, C) head-interleaved
  int wb = wh / NH, head = wh % NH;
  int n = lane & 15, mh = (lane >> 4) * 8;
#pragma unroll
  for (int r = 0; r < 8; ++r) {
    int tok = qt * 16 + r + mh;
    if (tok < NTOK) {
      size_t base = ((size_t)(wb * NTOK + tok)) * C_DIM + head * HD;
      Out[base + 0  + n] = f2bf(o0[r]);
      Out[base + 16 + n] = f2bf(o1[r]);
      Out[base + 32 + n] = f2bf(o2[r]);
      Out[base + 48 + n] = f2bf(o3[r]);
    }
  }
}

// ---------- K6: output projection (M=50176, N=768, K=768), 32x64 tile/wave ----------
__global__ __launch_bounds__(256) void proj_gemm_kernel(const unsigned short* __restrict__ A,
                                                        const unsigned short* __restrict__ W,
                                                        const float* __restrict__ bias,
                                                        float* __restrict__ out) {
  int wid  = (blockIdx.x * 256 + threadIdx.x) >> 5;
  int lane = threadIdx.x & 31;
  const int NT64 = C_DIM / 64;                 // 12
  const int MT   = MROWS / 32;                 // 1568
  if (wid >= MT * NT64) return;
  int mt = wid / NT64, nt = wid % NT64;

  const unsigned short* A0 = A + (size_t)mt * 32 * C_DIM;
  const unsigned short* A1 = A0 + (size_t)16 * C_DIM;
  const unsigned short* Bp = W + (size_t)nt * 64 * C_DIM;

  v8f acc[2][4];
#pragma unroll
  for (int mi = 0; mi < 2; ++mi)
#pragma unroll
    for (int t = 0; t < 4; ++t) acc[mi][t] = v8f{};

  v16bf a0 = load_a(A0, C_DIM), a1 = load_a(A1, C_DIM);
  v16bf b0 = load_bT(Bp + 0 * 16 * C_DIM, C_DIM);
  v16bf b1 = load_bT(Bp + 1 * 16 * C_DIM, C_DIM);
  v16bf b2 = load_bT(Bp + 2 * 16 * C_DIM, C_DIM);
  v16bf b3 = load_bT(Bp + 3 * 16 * C_DIM, C_DIM);

  for (int ks = 1; ks < KSTEPS; ++ks) {
    int k = ks * 32;
    v16bf a0n = load_a(A0 + k, C_DIM), a1n = load_a(A1 + k, C_DIM);
    v16bf b0n = load_bT(Bp + 0 * 16 * C_DIM + k, C_DIM);
    v16bf b1n = load_bT(Bp + 1 * 16 * C_DIM + k, C_DIM);
    v16bf b2n = load_bT(Bp + 2 * 16 * C_DIM + k, C_DIM);
    v16bf b3n = load_bT(Bp + 3 * 16 * C_DIM + k, C_DIM);
    acc[0][0] = wmma_bf16(a0, b0, acc[0][0]);
    acc[1][0] = wmma_bf16(a1, b0, acc[1][0]);
    acc[0][1] = wmma_bf16(a0, b1, acc[0][1]);
    acc[1][1] = wmma_bf16(a1, b1, acc[1][1]);
    acc[0][2] = wmma_bf16(a0, b2, acc[0][2]);
    acc[1][2] = wmma_bf16(a1, b2, acc[1][2]);
    acc[0][3] = wmma_bf16(a0, b3, acc[0][3]);
    acc[1][3] = wmma_bf16(a1, b3, acc[1][3]);
    a0 = a0n; a1 = a1n; b0 = b0n; b1 = b1n; b2 = b2n; b3 = b3n;
  }
  acc[0][0] = wmma_bf16(a0, b0, acc[0][0]);
  acc[1][0] = wmma_bf16(a1, b0, acc[1][0]);
  acc[0][1] = wmma_bf16(a0, b1, acc[0][1]);
  acc[1][1] = wmma_bf16(a1, b1, acc[1][1]);
  acc[0][2] = wmma_bf16(a0, b2, acc[0][2]);
  acc[1][2] = wmma_bf16(a1, b2, acc[1][2]);
  acc[0][3] = wmma_bf16(a0, b3, acc[0][3]);
  acc[1][3] = wmma_bf16(a1, b3, acc[1][3]);

  int nloc = lane & 15, mh = (lane >> 4) * 8;
#pragma unroll
  for (int mi = 0; mi < 2; ++mi) {
#pragma unroll
    for (int t = 0; t < 4; ++t) {
      int n = nt * 64 + t * 16 + nloc;
      float bv = bias[n];
#pragma unroll
      for (int r = 0; r < 8; ++r) {
        int m = mt * 32 + mi * 16 + r + mh;
        int wb = m / NTOK, tok = m % NTOK;
        int b = wb >> 4, wy = (wb >> 2) & 3, wx = wb & 3;
        int iy = tok / WSZ, ix = tok % WSZ;
        size_t dst = (((size_t)b * 56 + (wy * WSZ + iy)) * 56 + (wx * WSZ + ix)) * C_DIM + n;
        out[dst] = acc[mi][t][r] + bv;
      }
    }
  }
}

// ---------- host launch ----------
extern "C" void kernel_launch(void* const* d_in, const int* in_sizes, int n_in,
                              void* d_out, int out_size, void* d_ws, size_t ws_size,
                              hipStream_t stream) {
  const float* x      = (const float*)d_in[0];
  const float* qkv_w  = (const float*)d_in[1];
  const float* qkv_b  = (const float*)d_in[2];
  const float* proj_w = (const float*)d_in[3];
  const float* proj_b = (const float*)d_in[4];
  const float* rph    = (const float*)d_in[5];
  const float* rpw    = (const float*)d_in[6];
  float* out = (float*)d_out;

  // workspace carve-up (all bf16 buffers, 16B-aligned offsets)
  unsigned short* Wqkv  = (unsigned short*)d_ws;
  unsigned short* Wproj = Wqkv  + (size_t)QKV_N * C_DIM;
  unsigned short* Xw    = Wproj + (size_t)C_DIM * C_DIM;
  unsigned short* Qb    = Xw    + (size_t)MROWS * C_DIM;
  unsigned short* Kb    = Qb    + (size_t)WHTOT * NP * HD;
  unsigned short* Vb    = Kb    + (size_t)WHTOT * NP * HD;
  unsigned short* Ao    = Vb    + (size_t)WHTOT * HD * NP;

  {
    size_t tot = (size_t)QKV_N * C_DIM + (size_t)C_DIM * C_DIM;
    conv_w_kernel<<<(unsigned)((tot + 255) / 256), 256, 0, stream>>>(qkv_w, proj_w, Wqkv, Wproj);
  }
  {
    size_t tot = (size_t)MROWS * C_DIM;
    winpart_kernel<<<(unsigned)((tot + 255) / 256), 256, 0, stream>>>(x, Xw);
  }
  {
    int tot = WHTOT * (NP - NTOK) * HD;
    zeropad_kernel<<<(tot + 255) / 256, 256, 0, stream>>>(Qb, Kb, Vb);
  }
  {
    int waves = (MROWS / 32) * (QKV_N / 64);           // 56448
    qkv_gemm_kernel<<<(waves * 32 + 255) / 256, 256, 0, stream>>>(Xw, Wqkv, qkv_b, Qb, Kb, Vb);
  }
  {
    dim3 grid(WHTOT, (NP - 16) / 16);                  // 3072 x 13
    attn_kernel<<<grid, 32, 0, stream>>>(Qb, Kb, Vb, rph, rpw, Ao);
  }
  {
    int waves = (MROWS / 32) * (C_DIM / 64);           // 18816
    proj_gemm_kernel<<<(waves * 32 + 255) / 256, 256, 0, stream>>>(Ao, Wproj, proj_b, out);
  }
}
